// FeatureGraphLayer_41137196761318
// MI455X (gfx1250) — matrix operand order repaired and checked by reading, not compiled, not run on hardware
//
#include <hip/hip_runtime.h>
#include <hip/hip_bf16.h>

typedef __attribute__((ext_vector_type(16))) _Float16 v16h;
typedef __attribute__((ext_vector_type(8)))  _Float16 v8h;
typedef __attribute__((ext_vector_type(8)))  float    v8f;

#define THREADS 256
#define NWAVES  8
#define H       128

// ---------------- out = x (float4 copy) ----------------
__global__ void fgl_init_out(const float4* __restrict__ x, float4* __restrict__ out, int n4) {
    int i = blockIdx.x * blockDim.x + threadIdx.x;
    if (i < n4) out[i] = x[i];
}

// ---------------- per-edge MLP + scatter-add ----------------
// Dynamic LDS layout (bytes):
//   w1t : [128][256] f16  (W1 transposed, N-major)          65536
//   w2t : [128][128] f16  (W2 transposed, N-major)          32768
//   a2  : [NWAVES][16][128] f16 (relu(layer1) staging)      32768
//   dstb: [NWAVES][16] int                                    512
//   attb: [NWAVES][16] float                                  512
extern __shared__ char fgl_smem[];

__global__ __launch_bounds__(THREADS) void fgl_edge_mlp(
    const float* __restrict__ x,
    const int*   __restrict__ ei,    // [2][E]
    const float* __restrict__ attr,  // [E]
    const float* __restrict__ W1,    // [256][128] row-major (K,N)
    const float* __restrict__ b1,    // [128]
    const float* __restrict__ W2,    // [128][128]
    const float* __restrict__ b2,    // [128]
    float* __restrict__ out,         // [N][128]
    int E)
{
    _Float16* w1t    = (_Float16*)fgl_smem;                 // [128][256]
    _Float16* w2t    = w1t + 128 * 256;                     // [128][128]
    _Float16* a2base = w2t + 128 * 128;                     // [NWAVES][16][128]
    int*      dstb   = (int*)(a2base + NWAVES * 16 * 128);  // [NWAVES][16]
    float*    attb   = (float*)(dstb + NWAVES * 16);        // [NWAVES][16]

    const int tid = threadIdx.x;

    // Cooperative load: convert + transpose weights into LDS (f16, N-major)
    for (int i = tid; i < 256 * 128; i += THREADS) {
        int k = i >> 7, n = i & 127;
        w1t[n * 256 + k] = (_Float16)W1[i];
    }
    for (int i = tid; i < 128 * 128; i += THREADS) {
        int k = i >> 7, n = i & 127;
        w2t[n * 128 + k] = (_Float16)W2[i];
    }
    __syncthreads();

    const int lane  = tid & 31;
    const int wave  = tid >> 5;
    const int mrow  = lane & 15;   // A-frag row M / B-frag column N / C-frag column N
    const int khalf = lane >> 4;   // lane-half selector in K
    _Float16* a2    = a2base + wave * 16 * 128;
    int*      dstw  = dstb + wave * 16;
    float*    attw  = attb + wave * 16;

    // Bias values for C init: N = n*16 + mrow, uniform across M (all 8 C regs)
    float b1v[8], b2v[8];
#pragma unroll
    for (int n = 0; n < 8; ++n) {
        b1v[n] = b1[n * 16 + mrow];
        b2v[n] = b2[n * 16 + mrow];
    }

    const int ntiles  = (E + 15) >> 4;
    const int gwave   = blockIdx.x * NWAVES + wave;
    const int gstride = gridDim.x * NWAVES;

    for (int t = gwave; t < ntiles; t += gstride) {
        const int ebase = t * 16;

        // Stash per-tile edge metadata (dst row + attr; attr=0 kills tail edges)
        if (lane < 16) {
            int e   = ebase + lane;
            bool ok = (e < E);
            int ec  = ok ? e : 0;
            dstw[lane] = ei[E + ec];
            attw[lane] = ok ? attr[ec] : 0.0f;
        }

        // Per-lane edge for the A-matrix gather (row M = mrow)
        int eg = ebase + mrow; if (eg >= E) eg = E - 1;
        const float* srcP = x + (long)ei[eg] * H;
        const float* dstP = x + (long)ei[E + eg] * H;

        // ---------- Layer 1: [16 x 256] @ [256 x 128] ----------
        v8f acc[8];
#pragma unroll
        for (int n = 0; n < 8; ++n) {
            v8f c;
#pragma unroll
            for (int r = 0; r < 8; ++r) c[r] = b1v[n];
            acc[n] = c;
        }

#pragma unroll
        for (int ks = 0; ks < 8; ++ks) {
            // A-frag (16x32 f16): lane holds row M=mrow, two 8-element K runs
            int k0 = ks * 32 + khalf * 8;        // VGPRs 0..3
            int k1 = ks * 32 + 16 + khalf * 8;   // VGPRs 4..7
            const float* p0 = (k0 < H) ? (srcP + k0) : (dstP + (k0 - H));
            const float* p1 = (k1 < H) ? (srcP + k1) : (dstP + (k1 - H));
            float4 f0 = *(const float4*)(p0);
            float4 f1 = *(const float4*)(p0 + 4);
            float4 f2 = *(const float4*)(p1);
            float4 f3 = *(const float4*)(p1 + 4);
            v16h a;
            a[0]  = (_Float16)f0.x; a[1]  = (_Float16)f0.y;
            a[2]  = (_Float16)f0.z; a[3]  = (_Float16)f0.w;
            a[4]  = (_Float16)f1.x; a[5]  = (_Float16)f1.y;
            a[6]  = (_Float16)f1.z; a[7]  = (_Float16)f1.w;
            a[8]  = (_Float16)f2.x; a[9]  = (_Float16)f2.y;
            a[10] = (_Float16)f2.z; a[11] = (_Float16)f2.w;
            a[12] = (_Float16)f3.x; a[13] = (_Float16)f3.y;
            a[14] = (_Float16)f3.z; a[15] = (_Float16)f3.w;

#pragma unroll
            for (int n = 0; n < 8; ++n) {
                // B-frag (32x16 f16): lane column N = n*16+mrow, two 8-element K runs
                const _Float16* bp = w1t + (n * 16 + mrow) * 256 + ks * 32 + khalf * 8;
                v8h blo = *(const v8h*)bp;
                v8h bhi = *(const v8h*)(bp + 16);
                v16h b;
#pragma unroll
                for (int i = 0; i < 8; ++i) { b[i] = blo[i]; b[8 + i] = bhi[i]; }
                acc[n] = __builtin_amdgcn_wmma_f32_16x16x32_f16(
                    false, a, false, b, (short)0, acc[n], false, false);
            }
        }

        // ---------- ReLU -> f16 staging in LDS (A for layer 2) ----------
        // C/D layout: VGPR r, lane half -> M = r + 8*khalf ; N = n*16 + mrow
#pragma unroll
        for (int n = 0; n < 8; ++n) {
#pragma unroll
            for (int r = 0; r < 8; ++r) {
                float v = acc[n][r];
                v = v > 0.0f ? v : 0.0f;
                int M = r + 8 * khalf;
                a2[M * H + n * 16 + mrow] = (_Float16)v;
            }
        }

        // ---------- Layer 2: [16 x 128] @ [128 x 128] ----------
        v8f acc2[8];
#pragma unroll
        for (int n = 0; n < 8; ++n) {
            v8f c;
#pragma unroll
            for (int r = 0; r < 8; ++r) c[r] = b2v[n];
            acc2[n] = c;
        }

#pragma unroll
        for (int ks = 0; ks < 4; ++ks) {
            const _Float16* ap0 = a2 + mrow * H + ks * 32 + khalf * 8;
            const _Float16* ap1 = ap0 + 16;
            v8h alo = *(const v8h*)ap0;
            v8h ahi = *(const v8h*)ap1;
            v16h a;
#pragma unroll
            for (int i = 0; i < 8; ++i) { a[i] = alo[i]; a[8 + i] = ahi[i]; }

#pragma unroll
            for (int n = 0; n < 8; ++n) {
                const _Float16* bp = w2t + (n * 16 + mrow) * 128 + ks * 32 + khalf * 8;
                v8h blo = *(const v8h*)bp;
                v8h bhi = *(const v8h*)(bp + 16);
                v16h b;
#pragma unroll
                for (int i = 0; i < 8; ++i) { b[i] = blo[i]; b[8 + i] = bhi[i]; }
                acc2[n] = __builtin_amdgcn_wmma_f32_16x16x32_f16(
                    false, a, false, b, (short)0, acc2[n], false, false);
            }
        }

        // ---------- scale by edge_attr, scatter-add into out[dst] ----------
#pragma unroll
        for (int r = 0; r < 8; ++r) {
            int M    = r + 8 * khalf;
            float aw = attw[M];
            int drow = dstw[M];
            float* op = out + (long)drow * H + mrow;
#pragma unroll
            for (int n = 0; n < 8; ++n) {
                float v = acc2[n][r] * aw;
                __hip_atomic_fetch_add(op + n * 16, v,
                                       __ATOMIC_RELAXED, __HIP_MEMORY_SCOPE_AGENT);
            }
        }
    }
}

extern "C" void kernel_launch(void* const* d_in, const int* in_sizes, int n_in,
                              void* d_out, int out_size, void* d_ws, size_t ws_size,
                              hipStream_t stream) {
    const float* x    = (const float*)d_in[0];
    const int*   ei   = (const int*)  d_in[1];
    const float* attr = (const float*)d_in[2];
    const float* W1   = (const float*)d_in[3];
    const float* b1   = (const float*)d_in[4];
    const float* W2   = (const float*)d_in[5];
    const float* b2   = (const float*)d_in[6];
    float* out = (float*)d_out;

    const int E  = in_sizes[2];   // edge_attr flat count == #edges
    const int NH = out_size;      // N_NODES * H

    // out = x
    int n4 = NH >> 2;
    fgl_init_out<<<(n4 + 255) / 256, 256, 0, stream>>>(
        (const float4*)x, (float4*)out, n4);

    // edge MLP + scatter
    int ntiles = (E + 15) >> 4;
    int blocks = (ntiles + NWAVES - 1) / NWAVES;
    if (blocks > 640) blocks = 640;
    size_t shmem = (size_t)(128 * 256 + 128 * 128 + NWAVES * 16 * 128) * sizeof(_Float16)
                 + (size_t)NWAVES * 16 * (sizeof(int) + sizeof(float));
    fgl_edge_mlp<<<blocks, THREADS, shmem, stream>>>(
        x, ei, attr, W1, b1, W2, b2, out, E);
}
